// Classifier_56504589746195
// MI455X (gfx1250) — compile-verified, matrix operand order checked by reading, not run on hardware
//
#include <hip/hip_runtime.h>
#include <hip/hip_bf16.h>
#include <math.h>

// Problem constants (from reference)
#define NB    128
#define DIN   5000
#define DINP  5024          // DIN padded to multiple of 32 (zero-filled)
#define HH    512
#define NCL   8192
#define CSZ   64
#define TOPK  10
#define NCAND (TOPK * CSZ)  // 640

typedef __attribute__((ext_vector_type(16))) __bf16 v16bf;
typedef __attribute__((ext_vector_type(8)))  __bf16 v8bf;
typedef __attribute__((ext_vector_type(8)))  float  v8f;

__device__ __forceinline__ float sigmoidf_(float x) {
    return 1.f / (1.f + __expf(-x));
}

// Gather 16 bf16 K-elements for one lane in the ISA's 16-bit A/B VGPR layout:
// elements 0..7 = K+0..7, elements 8..15 = K+16..23 (lane>=16 base handled by caller).
// Both halves are contiguous 16B loads -> global_load_b128.
__device__ __forceinline__ v16bf ld_tile16(const __bf16* __restrict__ p) {
    v8bf lo = *(const v8bf*)(p);
    v8bf hi = *(const v8bf*)(p + 16);
    return __builtin_shufflevector(lo, hi, 0, 1, 2, 3, 4, 5, 6, 7,
                                           8, 9, 10, 11, 12, 13, 14, 15);
}

// -------------------------------------------------------------------------
// Prep A: convert X fp32 -> bf16, zero-pad K from 5000 to 5024.
// -------------------------------------------------------------------------
__global__ __launch_bounds__(256) void k_cvt_pad_x(
    const float* __restrict__ X, __bf16* __restrict__ Xb)
{
    int idx = blockIdx.x * 256 + threadIdx.x;     // over 128*5024
    if (idx >= NB * DINP) return;
    int r = idx / DINP;
    int k = idx - r * DINP;
    Xb[idx] = (k < DIN) ? (__bf16)X[(size_t)r * DIN + k] : (__bf16)0.f;
}

// -------------------------------------------------------------------------
// Prep B: tiled transpose + convert. src[K][N] fp32 row-major ->
// dst[N][Kp] bf16 row-major (zero pad K..Kp). 64x64 LDS tile, coalesced
// on both global sides. N must be a multiple of 64.
// -------------------------------------------------------------------------
__global__ __launch_bounds__(256) void k_transpose_cvt(
    const float* __restrict__ src, __bf16* __restrict__ dst,
    int K, int N, int Kp)
{
    __shared__ float tile[64][65];
    const int k0 = blockIdx.x * 64;
    const int n0 = blockIdx.y * 64;

    // Load 64(k) x 64(n): consecutive threads -> consecutive n (coalesced).
    const int ln = threadIdx.x & 63;
    const int lk = threadIdx.x >> 6;              // 0..3
#pragma unroll
    for (int i = 0; i < 16; ++i) {
        int kk = k0 + lk + i * 4;
        tile[lk + i * 4][ln] = (kk < K) ? src[(size_t)kk * N + (n0 + ln)] : 0.f;
    }
    __syncthreads();

    // Store 64(n) x 64(k): consecutive threads -> consecutive k (coalesced).
    const int sk = threadIdx.x & 63;
    const int sn = threadIdx.x >> 6;
#pragma unroll
    for (int i = 0; i < 16; ++i) {
        int nn = sn + i * 4;
        int kk = k0 + sk;
        if (kk < Kp)
            dst[(size_t)(n0 + nn) * Kp + kk] = (__bf16)tile[sk][nn];
    }
}

// -------------------------------------------------------------------------
// GEMM1: y = X[128,5000] @ W1[5000,512] + b1 ; h = relu(y)
// bf16 WMMA, fp32 accumulate. One wave -> 16x32 tile (1 A load, 2 WMMAs).
// 8 M-tiles x 16 waves = 128 waves. Block = 128 threads (4 waves).
// -------------------------------------------------------------------------
__global__ __launch_bounds__(128) void k_gemm1(
    const __bf16* __restrict__ Xb, const __bf16* __restrict__ W1T,
    const float* __restrict__ b1, float* __restrict__ y_out,
    float* __restrict__ h_f32, __bf16* __restrict__ h_bf16)
{
    const int lane  = threadIdx.x & 31;
    const int gw    = (blockIdx.x * blockDim.x + threadIdx.x) >> 5;
    const int mt    = gw >> 4;                    // 0..7
    const int nt2   = gw & 15;                    // 0..15, covers 32 cols
    const int kbase = (lane >> 4) << 3;           // 0 or 8

    const __bf16* pa  = Xb  + (size_t)(mt  * 16 + (lane & 15)) * DINP + kbase;
    const __bf16* pb0 = W1T + (size_t)(nt2 * 32 + (lane & 15)) * DINP + kbase;
    const __bf16* pb1 = pb0 + (size_t)16 * DINP;

    v8f acc0 = {}, acc1 = {};
    for (int k0 = 0; k0 < DINP; k0 += 32) {       // 157 uniform iterations
        v16bf a  = ld_tile16(pa  + k0);
        v16bf b0 = ld_tile16(pb0 + k0);
        v16bf b1v = ld_tile16(pb1 + k0);
        acc0 = __builtin_amdgcn_wmma_f32_16x16x32_bf16(
            false, a, false, b0, (short)0, acc0, false, false);
        acc1 = __builtin_amdgcn_wmma_f32_16x16x32_bf16(
            false, a, false, b1v, (short)0, acc1, false, false);
    }

    // Epilogue. C/D map: VGPR r, lane l -> M = r + 8*(l>=16), N = l&15.
    const int mb = mt * 16 + ((lane >> 4) << 3);
    const int n0 = nt2 * 32 + (lane & 15);
#pragma unroll
    for (int half = 0; half < 2; ++half) {
        const int n = n0 + half * 16;
        const float bias = b1[n];
        const v8f acc = half ? acc1 : acc0;
#pragma unroll
        for (int r = 0; r < 8; ++r) {
            int m   = mb + r;
            float y = acc[r] + bias;
            y_out [(size_t)m * HH + n] = y;        // pre-relu output #1
            float h = y > 0.f ? y : 0.f;
            h_f32 [(size_t)m * HH + n] = h;
            h_bf16[(size_t)m * HH + n] = (__bf16)h;
        }
    }
}

// -------------------------------------------------------------------------
// GEMM2: meta = h[128,512] @ Wm[512,8192] + bm
// One wave -> 16x32 tile. 8 x 256 = 2048 waves. Block = 128 threads.
// -------------------------------------------------------------------------
__global__ __launch_bounds__(128) void k_gemm2(
    const __bf16* __restrict__ hb, const __bf16* __restrict__ WmT,
    const float* __restrict__ bm, float* __restrict__ meta)
{
    const int lane  = threadIdx.x & 31;
    const int gw    = (blockIdx.x * blockDim.x + threadIdx.x) >> 5;
    const int mt    = gw >> 8;                    // 0..7
    const int nt2   = gw & 255;                   // 0..255
    const int kbase = (lane >> 4) << 3;

    const __bf16* pa  = hb  + (size_t)(mt  * 16 + (lane & 15)) * HH + kbase;
    const __bf16* pb0 = WmT + (size_t)(nt2 * 32 + (lane & 15)) * HH + kbase;
    const __bf16* pb1 = pb0 + (size_t)16 * HH;

    v8f acc0 = {}, acc1 = {};
#pragma unroll 4
    for (int k0 = 0; k0 < HH; k0 += 32) {         // 16 iterations
        v16bf a  = ld_tile16(pa  + k0);
        v16bf b0 = ld_tile16(pb0 + k0);
        v16bf b1v = ld_tile16(pb1 + k0);
        acc0 = __builtin_amdgcn_wmma_f32_16x16x32_bf16(
            false, a, false, b0, (short)0, acc0, false, false);
        acc1 = __builtin_amdgcn_wmma_f32_16x16x32_bf16(
            false, a, false, b1v, (short)0, acc1, false, false);
    }

    const int mb = mt * 16 + ((lane >> 4) << 3);
    const int n0 = nt2 * 32 + (lane & 15);
#pragma unroll
    for (int half = 0; half < 2; ++half) {
        const int n = n0 + half * 16;
        const float bias = bm[n];
        const v8f acc = half ? acc1 : acc0;
#pragma unroll
        for (int r = 0; r < 8; ++r)
            meta[(size_t)(mb + r) * NCL + n] = acc[r] + bias;
    }
}

// -------------------------------------------------------------------------
// Top-K per row: 10 sequential argmax passes over 8192 logits.
// One block (256 threads = 8 waves) per batch row.
// -------------------------------------------------------------------------
__global__ __launch_bounds__(256) void k_topk(
    float* __restrict__ meta, int* __restrict__ idx_out,
    float* __restrict__ score_out)
{
    const int b = blockIdx.x;
    float* row = meta + (size_t)b * NCL;
    __shared__ float smax[8];
    __shared__ int   simax[8];

    for (int t = 0; t < TOPK; ++t) {
        float best = -INFINITY;
        int   bi   = 0;
        for (int i = threadIdx.x; i < NCL; i += 256) {
            float v = row[i];
            if (v > best) { best = v; bi = i; }
        }
#pragma unroll
        for (int off = 16; off > 0; off >>= 1) {
            float ov = __shfl_down(best, off);
            int   oi = __shfl_down(bi, off);
            if (ov > best) { best = ov; bi = oi; }
        }
        const int w = threadIdx.x >> 5;
        if ((threadIdx.x & 31) == 0) { smax[w] = best; simax[w] = bi; }
        __syncthreads();
        if (threadIdx.x == 0) {
            float gb = smax[0]; int gi = simax[0];
#pragma unroll
            for (int i = 1; i < 8; ++i)
                if (smax[i] > gb) { gb = smax[i]; gi = simax[i]; }
            idx_out  [b * TOPK + t] = gi;
            score_out[b * TOPK + t] = sigmoidf_(gb);
            row[gi] = -INFINITY;
        }
        __syncthreads();
    }
}

// -------------------------------------------------------------------------
// Candidate expansion + gathered-embedding scoring (HBM-bound: ~167MB).
// One wave per candidate; coalesced float4 loads; wave-shfl reduce.
// -------------------------------------------------------------------------
__global__ __launch_bounds__(256) void k_score(
    const float* __restrict__ hf, const float* __restrict__ E,
    const int* __restrict__ cmap, const int* __restrict__ tidx,
    const float* __restrict__ tscore, float* __restrict__ cand_out,
    float* __restrict__ score_out)
{
    const int lane = threadIdx.x & 31;
    const int gw   = (blockIdx.x * blockDim.x + threadIdx.x) >> 5;
    const int b    = gw / NCAND;
    const int c    = gw - b * NCAND;
    const int kk   = c >> 6;
    const int cluster = tidx[b * TOPK + kk];
    const int label   = cmap[(size_t)cluster * CSZ + (c & 63)];

    const float4* pe = (const float4*)(E  + (size_t)label * HH);
    const float4* ph = (const float4*)(hf + (size_t)b * HH);
    float sum = 0.f;
#pragma unroll
    for (int i = 0; i < 4; ++i) {
        float4 e4 = pe[i * 32 + lane];
        float4 h4 = ph[i * 32 + lane];
        sum += e4.x * h4.x + e4.y * h4.y + e4.z * h4.z + e4.w * h4.w;
    }
#pragma unroll
    for (int off = 16; off > 0; off >>= 1)
        sum += __shfl_xor(sum, off);

    if (lane == 0) {
        cand_out [(size_t)b * NCAND + c] = (float)label;
        score_out[(size_t)b * NCAND + c] = sigmoidf_(sum) * tscore[b * TOPK + kk];
    }
}

// -------------------------------------------------------------------------
extern "C" void kernel_launch(void* const* d_in, const int* in_sizes, int n_in,
                              void* d_out, int out_size, void* d_ws, size_t ws_size,
                              hipStream_t stream)
{
    const float* X    = (const float*)d_in[0];   // [128,5000]
    const float* W1   = (const float*)d_in[1];   // [5000,512]
    const float* b1   = (const float*)d_in[2];   // [512]
    const float* Wm   = (const float*)d_in[3];   // [512,8192]
    const float* bm   = (const float*)d_in[4];   // [8192]
    const float* E    = (const float*)d_in[5];   // [524288,512]
    const int*   cmap = (const int*)  d_in[6];   // [8192,64]

    // Workspace layout (bytes)
    char* ws = (char*)d_ws;
    __bf16* Xb     = (__bf16*)(ws);                       //  1,286,144
    __bf16* W1T    = (__bf16*)(ws + 1286144);             //  5,144,576
    __bf16* WmT    = (__bf16*)(ws + 6430720);             //  8,388,608
    __bf16* hb     = (__bf16*)(ws + 14819328);            //    131,072
    float*  hf     = (float*) (ws + 14950400);            //    262,144
    float*  meta   = (float*) (ws + 15212544);            //  4,194,304
    int*    tidx   = (int*)   (ws + 19406848);            //      5,120
    float*  tscore = (float*) (ws + 19411968);            //      5,120

    // Output layout: y [128*512] | cand [128*640] | candidates_scores [128*640]
    float* y_out    = (float*)d_out;
    float* cand_out = y_out + (size_t)NB * HH;
    float* cs_out   = cand_out + (size_t)NB * NCAND;

    // Prep: bf16 conversion + WMMA-friendly layouts
    k_cvt_pad_x    <<<(NB * DINP + 255) / 256, 256, 0, stream>>>(X, Xb);
    k_transpose_cvt<<<dim3(79, 8),   256, 0, stream>>>(W1, W1T, DIN, HH,  DINP);
    k_transpose_cvt<<<dim3(8, 128),  256, 0, stream>>>(Wm, WmT, HH,  NCL, HH);

    // Main pipeline
    k_gemm1<<<32,    128, 0, stream>>>(Xb, W1T, b1, y_out, hf, hb);
    k_gemm2<<<512,   128, 0, stream>>>(hb, WmT, bm, meta);
    k_topk <<<NB,    256, 0, stream>>>(meta, tidx, tscore);
    k_score<<<10240, 256, 0, stream>>>(hf, E, cmap, tidx, tscore, cand_out, cs_out);
}
